// Attention_6717328851554
// MI455X (gfx1250) — matrix operand order repaired and checked by reading, not compiled, not run on hardware
//
#include <hip/hip_runtime.h>
#include <hip/hip_bf16.h>
#include <math.h>

// ---------------- problem constants ----------------
#define BATCH    2
#define SEQ      2049
#define SEQ_PAD  2080          // 65*32 : multiple of 32 (key blocks) and 16 (tiles)
#define NHEADS   16
#define DHEAD    64
#define D_MODEL  1024
#define D_QKV    3072
#define SCALE_QK 0.125f        // 64^-0.5
#define NEG_BIG  (-3.0e38f)

typedef __bf16 bf16;
typedef __attribute__((ext_vector_type(16))) bf16  v16bf;
typedef __attribute__((ext_vector_type(8)))  bf16  v8bf;
typedef __attribute__((ext_vector_type(8)))  float v8f;

#define WMMA_BF16(A, B, C) \
  __builtin_amdgcn_wmma_f32_16x16x32_bf16(false, (A), false, (B), (short)0, (C), false, false)

// ===================================================================
// Fragment loaders per CDNA5 ISA 7.12.2 (wave32, 16-bit data).
// A (16x32 MxK): lane l -> row m=l&15, half h=l>>4; values are two
//   contiguous K-runs: K = h*8+{0..7} and 16+h*8+{0..7}  -> 2x b128.
// B (32x16 KxN): lane l -> col n=l&15, half h=l>>4; values are two
//   contiguous K-runs K = h*8+{0..7} and 16+h*8+{0..7}. With the B operand stored
//   K-innermost (transposed weight / Kbuf / Vt) -> 2x b128.
// C/D (16x16 f32): VGPR r -> row m = r + 8*h, col n = l&15.
// ===================================================================
__device__ inline v16bf cat8(v8bf lo, v8bf hi) {
  return __builtin_shufflevector(lo, hi, 0, 1, 2, 3, 4, 5, 6, 7,
                                 8, 9, 10, 11, 12, 13, 14, 15);
}

__device__ inline v16bf load_a_frag(const bf16* __restrict__ A, int lda) {
  int l = threadIdx.x & 31, m = l & 15, lh = l >> 4;
  const bf16* row = A + (size_t)m * lda + lh * 8;
  return cat8(*(const v8bf*)row, *(const v8bf*)(row + 16));
}

// Bt is K-innermost: element (k, n) = Bt[n*ldk + k]
__device__ inline v16bf load_bt_frag(const bf16* __restrict__ Bt, int ldk) {
  int l = threadIdx.x & 31, n = l & 15, lh = l >> 4;
  const bf16* p = Bt + (size_t)n * ldk + lh * 8;
  return cat8(*(const v8bf*)p, *(const v8bf*)(p + 16));
}

// ===================================================================
// 1) LayerNorm -> bf16, rows padded with zeros to SEQ_PAD
// ===================================================================
__global__ __launch_bounds__(256) void ln_kernel(const float* __restrict__ x,
                                                 const float* __restrict__ gamma,
                                                 const float* __restrict__ beta,
                                                 bf16* __restrict__ xn) {
  int row = blockIdx.x;                   // 0 .. B*SEQ_PAD-1
  int b = row / SEQ_PAD, tok = row % SEQ_PAD;
  bf16* out = xn + (size_t)row * D_MODEL;
  if (tok >= SEQ) {
    for (int i = threadIdx.x; i < D_MODEL; i += 256) out[i] = (bf16)0.0f;
    return;
  }
  const float* in = x + ((size_t)b * SEQ + tok) * D_MODEL;
  float vals[4], lsum = 0.f, lsq = 0.f;
#pragma unroll
  for (int i = 0; i < 4; ++i) {
    float v = in[threadIdx.x + 256 * i];
    vals[i] = v; lsum += v; lsq += v * v;
  }
  __shared__ float s_sum[256], s_sq[256];
  s_sum[threadIdx.x] = lsum; s_sq[threadIdx.x] = lsq;
  __syncthreads();
  for (int off = 128; off > 0; off >>= 1) {
    if (threadIdx.x < off) {
      s_sum[threadIdx.x] += s_sum[threadIdx.x + off];
      s_sq[threadIdx.x]  += s_sq[threadIdx.x + off];
    }
    __syncthreads();
  }
  float mean = s_sum[0] * (1.0f / D_MODEL);
  float var  = s_sq[0] * (1.0f / D_MODEL) - mean * mean;
  float rstd = rsqrtf(var + 1e-5f);
#pragma unroll
  for (int i = 0; i < 4; ++i) {
    int idx = threadIdx.x + 256 * i;
    out[idx] = (bf16)((vals[i] - mean) * rstd * gamma[idx] + beta[idx]);
  }
}

// ===================================================================
// 2) fp32 [K,N] -> bf16 [N,K] transpose (LDS 32x32 tiles)
// ===================================================================
__global__ __launch_bounds__(256) void transpose_cvt_kernel(const float* __restrict__ W,
                                                            bf16* __restrict__ Wt,
                                                            int K, int N) {
  __shared__ float tile[32][33];
  int k0 = blockIdx.x * 32, n0 = blockIdx.y * 32;
  int tx = threadIdx.x & 31, ty = threadIdx.x >> 5;   // 8 rows per pass
#pragma unroll
  for (int i = 0; i < 4; ++i) {
    int r = ty + 8 * i;
    tile[r][tx] = W[(size_t)(k0 + r) * N + n0 + tx];
  }
  __syncthreads();
#pragma unroll
  for (int i = 0; i < 4; ++i) {
    int r = ty + 8 * i;
    Wt[(size_t)(n0 + r) * K + k0 + tx] = (bf16)tile[tx][r];
  }
}

// ===================================================================
// 3) GEMM core: one wave computes a 32x64 tile, 8 WMMAs per k-step
// ===================================================================
__device__ inline void mma_32x64(const bf16* __restrict__ A0, // 32 rows, lda=K
                                 const bf16* __restrict__ Bt, // 64 cols, K-innermost
                                 int K, v8f acc[2][4]) {
  for (int k0 = 0; k0 < K; k0 += 32) {
    __builtin_prefetch(Bt + k0 + 512, 0, 0);          // global_prefetch_b8
    v16bf a0 = load_a_frag(A0 + k0, K);
    v16bf a1 = load_a_frag(A0 + (size_t)16 * K + k0, K);
#pragma unroll
    for (int c = 0; c < 4; ++c) {
      v16bf b = load_bt_frag(Bt + (size_t)c * 16 * K + k0, K);
      acc[0][c] = WMMA_BF16(a0, b, acc[0][c]);
      acc[1][c] = WMMA_BF16(a1, b, acc[1][c]);
    }
  }
}

// QKV GEMM: [B*SEQ_PAD,1024]bf16 @ Wt[3072,1024] -> qkv f32
__global__ __launch_bounds__(128) void gemm_qkv_kernel(const bf16* __restrict__ A,
                                                       const bf16* __restrict__ Bt,
                                                       float* __restrict__ C) {
  int wave = (int)((blockIdx.x * blockDim.x + threadIdx.x) >> 5);
  const int CB = D_QKV / 64;                           // 48
  const int RB = BATCH * SEQ_PAD / 32;                 // 130
  int rb = wave / CB, cb = wave % CB;
  if (rb >= RB) return;
  v8f acc[2][4] = {};
  mma_32x64(A + (size_t)rb * 32 * D_MODEL, Bt + (size_t)cb * 64 * D_MODEL,
            D_MODEL, acc);
  int l = threadIdx.x & 31, n = l & 15, lh = l >> 4;
  float* c_base = C + (size_t)rb * 32 * D_QKV + cb * 64;
#pragma unroll
  for (int s = 0; s < 2; ++s)
#pragma unroll
    for (int c = 0; c < 4; ++c)
#pragma unroll
      for (int r = 0; r < 8; ++r)
        c_base[(size_t)(s * 16 + r + 8 * lh) * D_QKV + c * 16 + n] = acc[s][c][r];
}

// Output GEMM: attnout[B*SEQ_PAD,1024]bf16 @ Wt[1024,1024] -> d_out f32
__global__ __launch_bounds__(128) void gemm_out_kernel(const bf16* __restrict__ A,
                                                       const bf16* __restrict__ Bt,
                                                       float* __restrict__ out) {
  int wave = (int)((blockIdx.x * blockDim.x + threadIdx.x) >> 5);
  const int CB = D_MODEL / 64;                         // 16
  const int RB = BATCH * SEQ_PAD / 32;                 // 130
  int rb = wave / CB, cb = wave % CB;
  if (rb >= RB) return;
  v8f acc[2][4] = {};
  mma_32x64(A + (size_t)rb * 32 * D_MODEL, Bt + (size_t)cb * 64 * D_MODEL,
            D_MODEL, acc);
  int l = threadIdx.x & 31, n = l & 15, lh = l >> 4;
#pragma unroll
  for (int s = 0; s < 2; ++s)
#pragma unroll
    for (int r = 0; r < 8; ++r) {
      int g = rb * 32 + s * 16 + r + 8 * lh;           // flat padded row
      int b = g / SEQ_PAD, tok = g % SEQ_PAD;
      if (tok >= SEQ) continue;
      float* o = out + ((size_t)b * SEQ + tok) * D_MODEL + cb * 64 + n;
#pragma unroll
      for (int c = 0; c < 4; ++c) o[c * 16] = acc[s][c][r];
    }
}

// ===================================================================
// 4) RoPE (rotate-half, pos = token-1, cls/pad skipped) + transpose:
//    Q,K -> [B*H, SEQ_PAD, 64];  V -> transposed [B*H, 64, SEQ_PAD]
// ===================================================================
__global__ __launch_bounds__(256) void rope_kernel(const float* __restrict__ qkv,
                                                   bf16* __restrict__ Q,
                                                   bf16* __restrict__ K,
                                                   bf16* __restrict__ Vt) {
  int row = blockIdx.x;                   // B*SEQ_PAD
  int b = row / SEQ_PAD, tok = row % SEQ_PAD;
  const float* src = qkv + (size_t)row * D_QKV;
  bool do_rope = (tok >= 1) && (tok < SEQ);
  float p = (float)(tok - 1);
  for (int e = threadIdx.x; e < 1024; e += 256) {
    int hd = e >> 6, d = e & 63;
    size_t bh = (size_t)(b * NHEADS + hd);
    size_t dst_qk = (bh * SEQ_PAD + tok) * DHEAD + d;
    size_t dst_v  = (bh * DHEAD + d) * SEQ_PAD + tok;
    float qv = src[e], kv = src[e + 1024], vv = src[e + 2048];
    if (do_rope) {
      int i = d & 31;                     // freq index (concat layout)
      float freq = p * __powf(10000.f, -(float)(2 * i) * (1.0f / 64.0f));
      float c = __cosf(freq), s = __sinf(freq);
      int ep = (e & ~63) | (d ^ 32);      // rotate-half partner
      float sgn = (d < 32) ? -1.f : 1.f;
      qv = qv * c + sgn * src[ep] * s;
      kv = kv * c + sgn * src[ep + 1024] * s;
    }
    Q[dst_qk] = (bf16)qv; K[dst_qk] = (bf16)kv; Vt[dst_v] = (bf16)vv;
  }
}

// ===================================================================
// 5) Flash attention: one wave per (b,h,16-query tile); 32-key blocks
// ===================================================================
#define P_PITCH 36    // floats; 144B = multiple of 16B for ds_load_b128

__global__ __launch_bounds__(128) void flash_kernel(const bf16* __restrict__ Q,
                                                    const bf16* __restrict__ K,
                                                    const bf16* __restrict__ Vt,
                                                    bf16* __restrict__ O) {
  __shared__ float lds_p[4][16][P_PITCH];
  int wslot = threadIdx.x >> 5;
  int lane  = threadIdx.x & 31;
  int gw = blockIdx.x * 4 + wslot;
  const int QT = SEQ_PAD / 16;            // 130
  int qt = gw % QT, bh = gw / QT;
  if (bh >= BATCH * NHEADS) return;

  const bf16* qb  = Q  + ((size_t)bh * SEQ_PAD + (size_t)qt * 16) * DHEAD;
  const bf16* kb  = K  + (size_t)bh * SEQ_PAD * DHEAD;
  const bf16* vtb = Vt + (size_t)bh * DHEAD * SEQ_PAD;
  int m = lane & 15, lh = lane >> 4, n = lane & 15;

  v16bf aq0 = load_a_frag(qb, DHEAD);           // dh 0..31
  v16bf aq1 = load_a_frag(qb + 32, DHEAD);      // dh 32..63

  v8f acc0 = {}, acc1 = {}, acc2 = {}, acc3 = {};
  float mrow[8], lrow[8];
#pragma unroll
  for (int r = 0; r < 8; ++r) { mrow[r] = NEG_BIG; lrow[r] = 0.f; }

  float (*P)[P_PITCH] = lds_p[wslot];

  for (int k0 = 0; k0 < SEQ_PAD; k0 += 32) {
    // ---- S = Q K^T for 32 keys (two 16-key subtiles) ----
    v8f s0 = {}, s1 = {};
    s0 = WMMA_BF16(aq0, load_bt_frag(kb + (size_t)k0 * DHEAD, DHEAD), s0);
    s0 = WMMA_BF16(aq1, load_bt_frag(kb + (size_t)k0 * DHEAD + 32, DHEAD), s0);
    s1 = WMMA_BF16(aq0, load_bt_frag(kb + (size_t)(k0 + 16) * DHEAD, DHEAD), s1);
    s1 = WMMA_BF16(aq1, load_bt_frag(kb + (size_t)(k0 + 16) * DHEAD + 32, DHEAD), s1);

    // ---- scale + key mask ----
    bool mask0 = (k0 + n) >= SEQ;
    bool mask1 = (k0 + 16 + n) >= SEQ;
#pragma unroll
    for (int r = 0; r < 8; ++r) {
      s0[r] = mask0 ? NEG_BIG : s0[r] * SCALE_QK;
      s1[r] = mask1 ? NEG_BIG : s1[r] * SCALE_QK;
    }

    // ---- online softmax (rows live across the 16-lane half) ----
    float mnew[8], alpha[8];
#pragma unroll
    for (int r = 0; r < 8; ++r) {
      float mx = fmaxf(s0[r], s1[r]);
#pragma unroll
      for (int off = 1; off < 16; off <<= 1) mx = fmaxf(mx, __shfl_xor(mx, off, 16));
      mnew[r]  = fmaxf(mrow[r], mx);
      alpha[r] = __expf(mrow[r] - mnew[r]);
      mrow[r]  = mnew[r];
    }
#pragma unroll
    for (int r = 0; r < 8; ++r) {
      float p0 = __expf(s0[r] - mnew[r]);
      float p1 = __expf(s1[r] - mnew[r]);
      float rs = p0 + p1;
#pragma unroll
      for (int off = 1; off < 16; off <<= 1) rs += __shfl_xor(rs, off, 16);
      lrow[r] = lrow[r] * alpha[r] + rs;
      P[r + 8 * lh][n]      = p0;
      P[r + 8 * lh][16 + n] = p1;
      acc0[r] *= alpha[r]; acc1[r] *= alpha[r];
      acc2[r] *= alpha[r]; acc3[r] *= alpha[r];
    }

    // ---- LDS C-layout -> A-layout re-fragmentation (same-wave, in-order) ----
    asm volatile("s_wait_dscnt 0" ::: "memory");
    float4 f0 = *(const float4*)&P[m][lh * 8];
    float4 f1 = *(const float4*)&P[m][lh * 8 + 4];
    float4 f2 = *(const float4*)&P[m][16 + lh * 8];
    float4 f3 = *(const float4*)&P[m][16 + lh * 8 + 4];
    v16bf ap;
    ap[0] = (bf16)f0.x; ap[1] = (bf16)f0.y; ap[2] = (bf16)f0.z; ap[3] = (bf16)f0.w;
    ap[4] = (bf16)f1.x; ap[5] = (bf16)f1.y; ap[6] = (bf16)f1.z; ap[7] = (bf16)f1.w;
    ap[8]  = (bf16)f2.x; ap[9]  = (bf16)f2.y; ap[10] = (bf16)f2.z; ap[11] = (bf16)f2.w;
    ap[12] = (bf16)f3.x; ap[13] = (bf16)f3.y; ap[14] = (bf16)f3.z; ap[15] = (bf16)f3.w;

    // ---- O += P @ V (4 dh tiles of 16), Vt is key-innermost ----
    acc0 = WMMA_BF16(ap, load_bt_frag(vtb + (size_t)0  * SEQ_PAD + k0, SEQ_PAD), acc0);
    acc1 = WMMA_BF16(ap, load_bt_frag(vtb + (size_t)16 * SEQ_PAD + k0, SEQ_PAD), acc1);
    acc2 = WMMA_BF16(ap, load_bt_frag(vtb + (size_t)32 * SEQ_PAD + k0, SEQ_PAD), acc2);
    acc3 = WMMA_BF16(ap, load_bt_frag(vtb + (size_t)48 * SEQ_PAD + k0, SEQ_PAD), acc3);
  }

  // ---- normalize and store to [B*SEQ_PAD, 1024] bf16 (col = head*64+dh) ----
  int b = bh / NHEADS, head = bh % NHEADS;
#pragma unroll
  for (int r = 0; r < 8; ++r) {
    int tok = qt * 16 + r + 8 * lh;
    if (tok >= SEQ) continue;
    float inv = 1.0f / lrow[r];
    size_t base = ((size_t)b * SEQ_PAD + tok) * D_MODEL + (size_t)head * DHEAD;
    O[base + 0  + n] = (bf16)(acc0[r] * inv);
    O[base + 16 + n] = (bf16)(acc1[r] * inv);
    O[base + 32 + n] = (bf16)(acc2[r] * inv);
    O[base + 48 + n] = (bf16)(acc3[r] * inv);
  }
}

// ===================================================================
// launch
// ===================================================================
extern "C" void kernel_launch(void* const* d_in, const int* in_sizes, int n_in,
                              void* d_out, int out_size, void* d_ws, size_t ws_size,
                              hipStream_t stream) {
  const float* x     = (const float*)d_in[0];
  const float* gamma = (const float*)d_in[1];
  const float* beta  = (const float*)d_in[2];
  const float* wqkv  = (const float*)d_in[3];
  const float* wout  = (const float*)d_in[4];
  float* out = (float*)d_out;

  char* ws = (char*)d_ws;
  size_t off = 0;
  auto take = [&](size_t bytes) -> char* {
    char* p = ws + off;
    off = (off + bytes + 255) & ~(size_t)255;
    return p;
  };
  bf16*  xn_b    = (bf16*)take((size_t)BATCH * SEQ_PAD * D_MODEL * sizeof(bf16));
  bf16*  wqkv_t  = (bf16*)take((size_t)D_MODEL * D_QKV * sizeof(bf16));   // [3072,1024]
  bf16*  wout_t  = (bf16*)take((size_t)D_MODEL * D_MODEL * sizeof(bf16)); // [1024,1024]
  float* qkv_f   = (float*)take((size_t)BATCH * SEQ_PAD * D_QKV * sizeof(float));
  bf16*  q_b     = (bf16*)take((size_t)BATCH * NHEADS * SEQ_PAD * DHEAD * sizeof(bf16));
  bf16*  k_b     = (bf16*)take((size_t)BATCH * NHEADS * SEQ_PAD * DHEAD * sizeof(bf16));
  bf16*  vt_b    = (bf16*)take((size_t)BATCH * NHEADS * DHEAD * SEQ_PAD * sizeof(bf16));
  bf16*  attno_b = (bf16*)take((size_t)BATCH * SEQ_PAD * D_MODEL * sizeof(bf16));
  (void)ws_size;

  // 1) LayerNorm + pad
  ln_kernel<<<BATCH * SEQ_PAD, 256, 0, stream>>>(x, gamma, beta, xn_b);

  // 2) weight transpose+convert: W[K,N] f32 -> Wt[N,K] bf16
  transpose_cvt_kernel<<<dim3(D_MODEL / 32, D_QKV / 32), 256, 0, stream>>>(
      wqkv, wqkv_t, D_MODEL, D_QKV);
  transpose_cvt_kernel<<<dim3(D_MODEL / 32, D_MODEL / 32), 256, 0, stream>>>(
      wout, wout_t, D_MODEL, D_MODEL);

  // 3) QKV GEMM (32x64 tiles per wave)
  {
    int waves = (BATCH * SEQ_PAD / 32) * (D_QKV / 64);   // 130*48
    gemm_qkv_kernel<<<(waves + 3) / 4, 128, 0, stream>>>(xn_b, wqkv_t, qkv_f);
  }

  // 4) RoPE + head transpose (V stored key-innermost)
  rope_kernel<<<BATCH * SEQ_PAD, 256, 0, stream>>>(qkv_f, q_b, k_b, vt_b);

  // 5) flash attention
  {
    int waves = BATCH * NHEADS * (SEQ_PAD / 16);         // 32*130
    flash_kernel<<<(waves + 3) / 4, 128, 0, stream>>>(q_b, k_b, vt_b, attno_b);
  }

  // 6) output projection
  {
    int waves = (BATCH * SEQ_PAD / 32) * (D_MODEL / 64); // 130*16
    gemm_out_kernel<<<(waves + 3) / 4, 128, 0, stream>>>(attno_b, wout_t, out);
  }
}
